// AttentionGCNLayer_56607668961683
// MI455X (gfx1250) — matrix operand order repaired and checked by reading, not compiled, not run on hardware
//
#include <hip/hip_runtime.h>
#include <stdint.h>

typedef __attribute__((ext_vector_type(16))) __bf16 v16bf;
typedef __attribute__((ext_vector_type(8)))  float  v8f;

#define DEVINL static __device__ __forceinline__
#define USE_ASYNC_LDS 1

DEVINL uint16_t f2bf(float f) {
  union { float f; uint32_t u; } x; x.f = f;
  uint32_t r = x.u + 0x7FFFu + ((x.u >> 16) & 1u);   // round-to-nearest-even
  return (uint16_t)(r >> 16);
}
DEVINL float bf2f(uint16_t h) {
  union { uint32_t u; float f; } x; x.u = (uint32_t)h << 16;
  return x.f;
}

// gfx1250 async global->LDS copy (ASYNCcnt-tracked, no VGPR round trip)
DEVINL void async_lds_b128(const void* gaddr, const uint16_t* lds_ptr) {
#if USE_ASYNC_LDS
  uint32_t lds_off = (uint32_t)(uintptr_t)lds_ptr;   // low 32 bits = LDS offset
  asm volatile("global_load_async_to_lds_b128 %0, %1, off"
               :: "v"(lds_off), "v"(gaddr) : "memory");
#else
  uint4 v = *(const uint4*)gaddr;
  uint32_t* d = (uint32_t*)lds_ptr;
  d[0] = v.x; d[1] = v.y; d[2] = v.z; d[3] = v.w;
#endif
}
DEVINL void wait_async() {
#if USE_ASYNC_LDS
  asm volatile("s_wait_asynccnt 0x0" ::: "memory");
#endif
}

// ---------------- tiling config ----------------
static constexpr int TBM = 128, TBN = 128, TBK = 32, TBKP = 40; // padded LDS row

enum { AM_F32 = 0, AM_BF16 = 1 };
enum { BMODE_TN_F32 = 0, BMODE_NT_BF16 = 1, BMODE_TN_BF16 = 2 };
enum { EPI_BIAS_BF16 = 0, EPI_SCALE_MASK_BF16 = 1, EPI_PLAIN_BF16 = 2,
       EPI_BIAS_F32 = 3, EPI_GCN = 4 };

struct GemmArgs {
  const void* A; int lda; long long aHi, aLo;   // strides in elements of A's type
  const void* B; int ldb; long long bHi, bLo;
  void*       C; int ldc; long long cHi, cLo;
  const float* bias;
  const float* degs; long long dHi, dLo;
  float*    X;  int ldx;  long long xHi,  xLo;  // in-place residual state (EPI_GCN)
  uint16_t* XB; int ldxb; long long xbHi, xbLo; // bf16 shadow / concat target (EPI_GCN)
  float alpha;
  int M, N, K, Hsplit;                          // batch z -> (b = z/Hsplit, h = z%Hsplit)
};

// A fragment: 16x32 bf16, lanes 0-15 hold K={0..7,16..23}, lanes 16-31 K={8..15,24..31}
DEVINL v16bf ld_fragA(const uint16_t (*S)[TBKP], int rowbase, int lane) {
  union { v16bf v; uint32_t w[8]; } u;
  const int m  = lane & 15;
  const int kh = (lane >> 4) * 8;
  const uint32_t* rp = (const uint32_t*)&S[rowbase + m][0];
  #pragma unroll
  for (int i = 0; i < 8; i++) {
    const int k = (i < 4) ? (kh + 2 * i) : (16 + kh + 2 * (i - 4));
    u.w[i] = rp[k >> 1];
  }
  return u.v;
}
// B fragment: 32x16 bf16 (stored transposed in LDS as [n][k]):
// lanes 0-15 hold K={0..7,16..23} (2 per VGPR), lanes 16-31 hold K={8..15,24..31}
DEVINL v16bf ld_fragB(const uint16_t (*S)[TBKP], int rowbase, int lane) {
  union { v16bf v; uint32_t w[8]; } u;
  const int n  = lane & 15;
  const int k8 = (lane >> 4) * 8;               // bf16 k offset (u32 offset = k8 >> 1)
  const uint32_t* rp = (const uint32_t*)&S[rowbase + n][0];
  #pragma unroll
  for (int i = 0; i < 8; i++) u.w[i] = rp[(i < 4) ? ((k8 >> 1) + i) : (8 + (k8 >> 1) + (i - 4))];
  return u.v;
}

// registers holding an in-flight staged tile (only mode-relevant members live)
struct StageRegs {
  float4 af[4];   // AM_F32
  float4 bf[4];   // BMODE_TN_F32
  uint4  b16[2];  // BMODE_TN_BF16
};

// Phase 1: issue global traffic (async for pure bf16 copies, regs otherwise)
template<int AMODE, int BMODE>
DEVINL void stage_load(const GemmArgs& g, StageRegs& sr,
                       uint16_t (*Asn)[TBKP], uint16_t (*Bsn)[TBKP],
                       const float* Apf, const uint16_t* Ap16,
                       const float* Bpf, const uint16_t* Bp16,
                       int row0, int col0, int k0, int tid) {
  if constexpr (AMODE == AM_F32) {
    #pragma unroll
    for (int it = 0; it < 4; it++) {
      const int ch = tid + it * 256;                 // 1024 float4 chunks
      const int r  = ch >> 3, c4 = (ch & 7) * 4;
      sr.af[it] = *(const float4*)(Apf + (size_t)(row0 + r) * g.lda + k0 + c4);
    }
  } else {                                           // bf16 A: async straight to LDS
    #pragma unroll
    for (int it = 0; it < 2; it++) {
      const int ch = tid + it * 256;                 // 512 uint4 chunks (8 bf16)
      const int r  = ch >> 2, c8 = (ch & 3) * 8;
      async_lds_b128(Ap16 + (size_t)(row0 + r) * g.lda + k0 + c8, &Asn[r][c8]);
    }
  }
  if constexpr (BMODE == BMODE_NT_BF16) {            // bf16 NT B: async copy
    #pragma unroll
    for (int it = 0; it < 2; it++) {
      const int ch = tid + it * 256;
      const int r  = ch >> 2, c8 = (ch & 3) * 8;
      async_lds_b128(Bp16 + (size_t)(col0 + r) * g.ldb + k0 + c8, &Bsn[r][c8]);
    }
  } else if constexpr (BMODE == BMODE_TN_F32) {
    #pragma unroll
    for (int it = 0; it < 4; it++) {
      const int ch = tid + it * 256;
      const int kr = ch >> 5, c4 = (ch & 31) * 4;
      sr.bf[it] = *(const float4*)(Bpf + (size_t)(k0 + kr) * g.ldb + col0 + c4);
    }
  } else {                                           // TN_BF16
    #pragma unroll
    for (int it = 0; it < 2; it++) {
      const int ch = tid + it * 256;
      const int kr = ch >> 4, c8 = (ch & 15) * 8;
      sr.b16[it] = *(const uint4*)(Bp16 + (size_t)(k0 + kr) * g.ldb + col0 + c8);
    }
  }
}

// Phase 2: convert/transpose register-staged data into LDS (after compute)
template<int AMODE, int BMODE>
DEVINL void stage_store(const StageRegs& sr,
                        uint16_t (*Asn)[TBKP], uint16_t (*Bsn)[TBKP], int tid) {
  if constexpr (AMODE == AM_F32) {
    #pragma unroll
    for (int it = 0; it < 4; it++) {
      const int ch = tid + it * 256;
      const int r  = ch >> 3, c4 = (ch & 7) * 4;
      uint32_t* d = (uint32_t*)&Asn[r][c4];
      d[0] = (uint32_t)f2bf(sr.af[it].x) | ((uint32_t)f2bf(sr.af[it].y) << 16);
      d[1] = (uint32_t)f2bf(sr.af[it].z) | ((uint32_t)f2bf(sr.af[it].w) << 16);
    }
  }
  if constexpr (BMODE == BMODE_TN_F32) {
    #pragma unroll
    for (int it = 0; it < 4; it++) {
      const int ch = tid + it * 256;
      const int kr = ch >> 5, c4 = (ch & 31) * 4;
      Bsn[c4 + 0][kr] = f2bf(sr.bf[it].x);
      Bsn[c4 + 1][kr] = f2bf(sr.bf[it].y);
      Bsn[c4 + 2][kr] = f2bf(sr.bf[it].z);
      Bsn[c4 + 3][kr] = f2bf(sr.bf[it].w);
    }
  } else if constexpr (BMODE == BMODE_TN_BF16) {
    #pragma unroll
    for (int it = 0; it < 2; it++) {
      const int ch = tid + it * 256;
      const int kr = ch >> 4, c8 = (ch & 15) * 8;
      const uint32_t w[4] = {sr.b16[it].x, sr.b16[it].y, sr.b16[it].z, sr.b16[it].w};
      #pragma unroll
      for (int j = 0; j < 4; j++) {
        Bsn[c8 + 2*j    ][kr] = (uint16_t)(w[j] & 0xFFFFu);
        Bsn[c8 + 2*j + 1][kr] = (uint16_t)(w[j] >> 16);
      }
    }
  }
}

DEVINL void compute_stage(const uint16_t (*As)[TBKP], const uint16_t (*Bs)[TBKP],
                          v8f (&acc)[2][4], int wrow, int wcol, int lane) {
  v16bf af[2], bfr[4];
  #pragma unroll
  for (int ms = 0; ms < 2; ms++) af[ms] = ld_fragA(As, wrow + ms * 16, lane);
  #pragma unroll
  for (int ns = 0; ns < 4; ns++) bfr[ns] = ld_fragB(Bs, wcol + ns * 16, lane);
  #pragma unroll
  for (int ms = 0; ms < 2; ms++)
    #pragma unroll
    for (int ns = 0; ns < 4; ns++)
      acc[ms][ns] = __builtin_amdgcn_wmma_f32_16x16x32_bf16(
          false, af[ms], false, bfr[ns], (short)0, acc[ms][ns], false, false);
}

template<int AMODE, int BMODE, int EPI>
__global__ __launch_bounds__(256) void gemm_wmma(GemmArgs g) {
  __shared__ uint16_t As[2][TBM][TBKP];   // double-buffered
  __shared__ uint16_t Bs[2][TBN][TBKP];   // stored transposed: [ncol][k]

  const int tid  = threadIdx.x;
  const int lane = tid & 31;
  const int wid  = tid >> 5;
  const int wrow = (wid & 3) * 32;              // 4 waves along M
  const int wcol = (wid >> 2) * 64;             // 2 waves along N

  const int z  = blockIdx.z;
  const int bb = z / g.Hsplit;
  const int hh = z % g.Hsplit;
  const int row0 = blockIdx.y * TBM;
  const int col0 = blockIdx.x * TBN;

  const float*    Apf  = (const float*)g.A    + (size_t)bb * g.aHi + (size_t)hh * g.aLo;
  const uint16_t* Ap16 = (const uint16_t*)g.A + (size_t)bb * g.aHi + (size_t)hh * g.aLo;
  const float*    Bpf  = (const float*)g.B    + (size_t)bb * g.bHi + (size_t)hh * g.bLo;
  const uint16_t* Bp16 = (const uint16_t*)g.B + (size_t)bb * g.bHi + (size_t)hh * g.bLo;

  constexpr bool useAsync = (AMODE == AM_BF16) || (BMODE == BMODE_NT_BF16);

  v8f acc[2][4];
  {
    v8f zero = {0.f,0.f,0.f,0.f,0.f,0.f,0.f,0.f};
    #pragma unroll
    for (int i = 0; i < 2; i++)
      #pragma unroll
      for (int j = 0; j < 4; j++) acc[i][j] = zero;
  }

  // ---- pipelined main loop: prefetch stage k+1 while computing stage k ----
  {
    StageRegs sr;
    stage_load<AMODE, BMODE>(g, sr, As[0], Bs[0], Apf, Ap16, Bpf, Bp16,
                             row0, col0, 0, tid);
    stage_store<AMODE, BMODE>(sr, As[0], Bs[0], tid);
    if (useAsync) wait_async();
    __syncthreads();
  }
  int cur = 0;
  for (int k0 = 0; k0 < g.K; k0 += TBK) {
    const int knext = k0 + TBK;
    StageRegs sr;
    if (knext < g.K)
      stage_load<AMODE, BMODE>(g, sr, As[1 - cur], Bs[1 - cur], Apf, Ap16, Bpf, Bp16,
                               row0, col0, knext, tid);
    compute_stage(As[cur], Bs[cur], acc, wrow, wcol, lane);
    if (knext < g.K)
      stage_store<AMODE, BMODE>(sr, As[1 - cur], Bs[1 - cur], tid);
    if (useAsync) wait_async();
    __syncthreads();
    cur ^= 1;
  }

  // ---------------- epilogue ----------------
  uint16_t* Cb = g.C ? ((uint16_t*)g.C + (size_t)bb * g.cHi + (size_t)hh * g.cLo) : nullptr;
  float*    Cf = g.C ? ((float*)g.C    + (size_t)bb * g.cHi + (size_t)hh * g.cLo) : nullptr;
  const float* degp = g.degs ? (g.degs + (size_t)bb * g.dHi + (size_t)hh * g.dLo) : nullptr;
  float*    Xp  = g.X  ? (g.X  + (size_t)bb * g.xHi  + (size_t)hh * g.xLo ) : nullptr;
  uint16_t* XBp = g.XB ? (g.XB + (size_t)bb * g.xbHi + (size_t)hh * g.xbLo) : nullptr;

  #pragma unroll
  for (int ms = 0; ms < 2; ms++) {
    #pragma unroll
    for (int ns = 0; ns < 4; ns++) {
      const int col   = col0 + wcol + ns * 16 + (lane & 15);
      const int rbase = row0 + wrow + ms * 16 + ((lane >> 4) * 8);
      #pragma unroll
      for (int r = 0; r < 8; r++) {
        const int row = rbase + r;
        const float a = acc[ms][ns][r];
        if constexpr (EPI == EPI_BIAS_BF16) {
          Cb[(size_t)row * g.ldc + col] = f2bf(a + g.bias[col]);
        } else if constexpr (EPI == EPI_SCALE_MASK_BF16) {
          float v = a * g.alpha;
          if (v == 0.0f) v = -1e9f;                       // masked_fill(scores==0)
          Cb[(size_t)row * g.ldc + col] = f2bf(v);
        } else if constexpr (EPI == EPI_PLAIN_BF16) {
          Cb[(size_t)row * g.ldc + col] = f2bf(a);
        } else if constexpr (EPI == EPI_BIAS_F32) {
          Cf[(size_t)row * g.ldc + col] = a + g.bias[col];
        } else { // EPI_GCN: x += relu(acc/deg + bias); write f32 + bf16 shadow
          const float deg = degp[row];
          float v = a / deg + g.bias[col];
          v = v > 0.0f ? v : 0.0f;
          const size_t xi = (size_t)row * g.ldx + col;
          const float xn = Xp[xi] + v;
          Xp[xi] = xn;
          XBp[(size_t)row * g.ldxb + col] = f2bf(xn);
        }
      }
    }
  }
}

// ---------------- in-place bf16 row softmax (ncols == 4096) ----------------
__global__ __launch_bounds__(256) void softmax_rows(uint16_t* attn, float* degs, int ncols) {
  __shared__ float red[256];
  const int tid = threadIdx.x;
  uint16_t* rp = attn + (size_t)blockIdx.x * ncols;
  float v[16];
  #pragma unroll
  for (int it = 0; it < 2; it++) {
    const uint4 q = ((const uint4*)rp)[tid + it * 256];
    const uint32_t w[4] = {q.x, q.y, q.z, q.w};
    #pragma unroll
    for (int j = 0; j < 4; j++) {
      v[it*8 + 2*j    ] = bf2f((uint16_t)(w[j] & 0xFFFFu));
      v[it*8 + 2*j + 1] = bf2f((uint16_t)(w[j] >> 16));
    }
  }
  float mx = -3.0e38f;
  #pragma unroll
  for (int i = 0; i < 16; i++) mx = fmaxf(mx, v[i]);
  red[tid] = mx; __syncthreads();
  for (int s = 128; s > 0; s >>= 1) { if (tid < s) red[tid] = fmaxf(red[tid], red[tid + s]); __syncthreads(); }
  mx = red[0]; __syncthreads();
  float sum = 0.f;
  #pragma unroll
  for (int i = 0; i < 16; i++) { v[i] = __expf(v[i] - mx); sum += v[i]; }
  red[tid] = sum; __syncthreads();
  for (int s = 128; s > 0; s >>= 1) { if (tid < s) red[tid] += red[tid + s]; __syncthreads(); }
  const float inv = 1.0f / red[0]; __syncthreads();
  float dsum = 0.f;
  uint16_t pb[16];
  #pragma unroll
  for (int i = 0; i < 16; i++) { pb[i] = f2bf(v[i] * inv); dsum += bf2f(pb[i]); }
  #pragma unroll
  for (int it = 0; it < 2; it++) {
    uint4 q;
    q.x = (uint32_t)pb[it*8+0] | ((uint32_t)pb[it*8+1] << 16);
    q.y = (uint32_t)pb[it*8+2] | ((uint32_t)pb[it*8+3] << 16);
    q.z = (uint32_t)pb[it*8+4] | ((uint32_t)pb[it*8+5] << 16);
    q.w = (uint32_t)pb[it*8+6] | ((uint32_t)pb[it*8+7] << 16);
    ((uint4*)rp)[tid + it * 256] = q;
  }
  red[tid] = dsum; __syncthreads();
  for (int s = 128; s > 0; s >>= 1) { if (tid < s) red[tid] += red[tid + s]; __syncthreads(); }
  if (tid == 0) degs[blockIdx.x] = red[0];   // degree normalization (~1.0, kept faithful)
}

// ---------------- broadcast nodes_embed over heads: x f32 + bf16 shadow ----------------
__global__ __launch_bounds__(256) void init_x(const float* __restrict__ ne,
                                              float* __restrict__ x,
                                              uint16_t* __restrict__ xb, int ND, int H) {
  const size_t i4  = ((size_t)blockIdx.x * 256 + threadIdx.x) * 4;
  const size_t bh  = i4 / (size_t)ND;
  const size_t b   = bh / (size_t)H;
  const size_t rem = i4 - bh * (size_t)ND;
  const float4 vv = *(const float4*)(ne + b * (size_t)ND + rem);
  *(float4*)(x + i4) = vv;
  uint2 p;
  p.x = (uint32_t)f2bf(vv.x) | ((uint32_t)f2bf(vv.y) << 16);
  p.y = (uint32_t)f2bf(vv.z) | ((uint32_t)f2bf(vv.w) << 16);
  *(uint2*)(xb + i4) = p;
}

extern "C" void kernel_launch(void* const* d_in, const int* in_sizes, int n_in,
                              void* d_out, int out_size, void* d_ws, size_t ws_size,
                              hipStream_t stream) {
  (void)in_sizes; (void)n_in; (void)out_size; (void)ws_size;
  constexpr int Bb = 2, Nn = 4096, Dd = 256, Hh = 2, DKk = 128, ITERS = 2;

  const float* nodes = (const float*)d_in[0];
  const float* WQw = (const float*)d_in[1];
  const float* WQb = (const float*)d_in[2];
  const float* WKw = (const float*)d_in[3];
  const float* WKb = (const float*)d_in[4];
  const float* gcnW = (const float*)d_in[5];
  const float* gcnB = (const float*)d_in[6];
  const float* aggW = (const float*)d_in[7];
  const float* aggB = (const float*)d_in[8];
  float* out = (float*)d_out;

  char* ws = (char*)d_ws;
  size_t off = 0;
  auto wsalloc = [&](size_t bytes) -> void* {
    void* p = ws + off; off += (bytes + 255) & ~(size_t)255; return p;
  };
  uint16_t* q    = (uint16_t*)wsalloc((size_t)Bb * Nn * Dd * 2);        // 4 MB
  uint16_t* kmat = (uint16_t*)wsalloc((size_t)Bb * Nn * Dd * 2);        // 4 MB
  uint16_t* attn = (uint16_t*)wsalloc((size_t)Bb * Hh * Nn * Nn * 2);   // 134 MB
  float*    x    = (float*)   wsalloc((size_t)Bb * Hh * Nn * Dd * 4);   // 16.8 MB
  uint16_t* xb   = (uint16_t*)wsalloc((size_t)Bb * Hh * Nn * Dd * 2);   // 8.4 MB
  uint16_t* hbuf = (uint16_t*)wsalloc((size_t)Bb * Hh * Nn * Dd * 2);   // 8.4 MB
  uint16_t* xcat = (uint16_t*)wsalloc((size_t)Bb * Nn * Hh * Dd * 2);   // 8.4 MB
  float*    degs = (float*)   wsalloc((size_t)Bb * Hh * Nn * 4);

  // 1. x = broadcast(nodes_embed) over heads (f32 + bf16)
  init_x<<<dim3((Bb * Hh * Nn * Dd) / 1024), dim3(256), 0, stream>>>(nodes, x, xb, Nn * Dd, Hh);

  // 2-3. q/k projections: bf16( nodes @ W + b )
  for (int which = 0; which < 2; which++) {
    GemmArgs g{};
    g.A = nodes; g.lda = Dd; g.aHi = (long long)Nn * Dd; g.aLo = 0;
    g.B = which ? WKw : WQw; g.ldb = Dd; g.bHi = 0; g.bLo = 0;
    g.C = which ? kmat : q;  g.ldc = Dd; g.cHi = (long long)Nn * Dd; g.cLo = 0;
    g.bias = which ? WKb : WQb;
    g.M = Nn; g.N = Dd; g.K = Dd; g.Hsplit = 1;
    gemm_wmma<AM_F32, BMODE_TN_F32, EPI_BIAS_BF16>
        <<<dim3(Dd / TBN, Nn / TBM, Bb), dim3(256), 0, stream>>>(g);
  }

  // 4. scores = q @ k^T / sqrt(dk), mask(==0 -> -1e9), bf16 into attn buffer
  {
    GemmArgs g{};
    g.A = q;    g.lda = Dd; g.aHi = (long long)Nn * Dd; g.aLo = DKk;
    g.B = kmat; g.ldb = Dd; g.bHi = (long long)Nn * Dd; g.bLo = DKk;   // NT: rows are k-vectors
    g.C = attn; g.ldc = Nn; g.cHi = (long long)Hh * Nn * Nn; g.cLo = (long long)Nn * Nn;
    g.alpha = 0.08838834764831845f;  // 1/sqrt(128)
    g.M = Nn; g.N = Nn; g.K = DKk; g.Hsplit = Hh;
    gemm_wmma<AM_BF16, BMODE_NT_BF16, EPI_SCALE_MASK_BF16>
        <<<dim3(Nn / TBN, Nn / TBM, Bb * Hh), dim3(256), 0, stream>>>(g);
  }

  // 5. softmax rows in place (+ degs)
  softmax_rows<<<dim3(Bb * Hh * Nn), dim3(256), 0, stream>>>(attn, degs, Nn);

  // 6. GCN iterations
  for (int it = 0; it < ITERS; it++) {
    { // h = bf16( xb @ gcn_W[it] )
      GemmArgs g{};
      g.A = xb; g.lda = Dd; g.aHi = (long long)Hh * Nn * Dd; g.aLo = (long long)Nn * Dd;
      g.B = gcnW + (size_t)it * Dd * Dd; g.ldb = Dd; g.bHi = 0; g.bLo = 0;
      g.C = hbuf; g.ldc = Dd; g.cHi = (long long)Hh * Nn * Dd; g.cLo = (long long)Nn * Dd;
      g.M = Nn; g.N = Dd; g.K = Dd; g.Hsplit = Hh;
      gemm_wmma<AM_BF16, BMODE_TN_F32, EPI_PLAIN_BF16>
          <<<dim3(Dd / TBN, Nn / TBM, Bb * Hh), dim3(256), 0, stream>>>(g);
    }
    { // x += relu( (attn@h)/degs + gcn_b[it] ); fused bf16 output for next stage
      GemmArgs g{};
      g.A = attn; g.lda = Nn; g.aHi = (long long)Hh * Nn * Nn; g.aLo = (long long)Nn * Nn;
      g.B = hbuf; g.ldb = Dd; g.bHi = (long long)Hh * Nn * Dd; g.bLo = (long long)Nn * Dd;
      g.bias = gcnB + (size_t)it * Dd;
      g.degs = degs; g.dHi = (long long)Hh * Nn; g.dLo = Nn;
      g.X = x; g.ldx = Dd; g.xHi = (long long)Hh * Nn * Dd; g.xLo = (long long)Nn * Dd;
      if (it < ITERS - 1) {  // -> xb for next gcn_W matmul
        g.XB = xb; g.ldxb = Dd;
        g.xbHi = (long long)Hh * Nn * Dd; g.xbLo = (long long)Nn * Dd;
      } else {               // -> head-concat layout [B][N][H*D]
        g.XB = xcat; g.ldxb = Hh * Dd;
        g.xbHi = (long long)Nn * Hh * Dd; g.xbLo = Dd;
      }
      g.M = Nn; g.N = Dd; g.K = Nn; g.Hsplit = Hh;
      gemm_wmma<AM_BF16, BMODE_TN_BF16, EPI_GCN>
          <<<dim3(Dd / TBN, Nn / TBM, Bb * Hh), dim3(256), 0, stream>>>(g);
    }
  }

  // 7. out = xcat @ agg_W + agg_b (f32 output)
  {
    GemmArgs g{};
    g.A = xcat; g.lda = Hh * Dd; g.aHi = (long long)Nn * Hh * Dd; g.aLo = 0;
    g.B = aggW; g.ldb = Dd; g.bHi = 0; g.bLo = 0;
    g.C = out;  g.ldc = Dd; g.cHi = (long long)Nn * Dd; g.cLo = 0;
    g.bias = aggB;
    g.M = Nn; g.N = Dd; g.K = Hh * Dd; g.Hsplit = 1;
    gemm_wmma<AM_BF16, BMODE_TN_F32, EPI_BIAS_F32>
        <<<dim3(Dd / TBN, Nn / TBM, Bb), dim3(256), 0, stream>>>(g);
  }
}
